// Embedding_54357106098462
// MI455X (gfx1250) — matrix operand order repaired and checked by reading, not compiled, not run on hardware
//
#include <hip/hip_runtime.h>

#define N_ROWS 65536
#define DIM 256
#define KCODES 512
#define ROWS_PER_BLOCK 128
#define LDS_STRIDE 264  // 256 + 8 bf16 pad -> lane bank stride 132 dwords == 4 mod 64

typedef __attribute__((ext_vector_type(16))) __bf16        v16bf;
typedef __attribute__((ext_vector_type(8)))  float         v8f;
typedef __attribute__((ext_vector_type(4)))  unsigned int  u32x4;
typedef __attribute__((ext_vector_type(4)))  float         f32x4;
typedef __attribute__((ext_vector_type(2)))  float         f32x2;

union FragAB { u32x4 q[2]; v16bf v; };

__device__ inline unsigned short f2bf(float f) {
  unsigned int u = __float_as_uint(f);
  u += 0x7fffu + ((u >> 16) & 1u);      // round-to-nearest-even
  return (unsigned short)(u >> 16);
}

__global__ void vq_zero_loss(float* out) { out[0] = 0.0f; }

__global__ __launch_bounds__(256)
void vq_main(const float* __restrict__ x, const float* __restrict__ weight,
             float* __restrict__ out)
{
  extern __shared__ unsigned char smem_raw[];
  unsigned short* xa = (unsigned short*)smem_raw;                    // 128*264 bf16
  unsigned short* wb = xa + ROWS_PER_BLOCK * LDS_STRIDE;             // 128*264 bf16
  float*          wn = (float*)(wb + ROWS_PER_BLOCK * LDS_STRIDE);   // 128 f32
  int*           ids = (int*)(wn + ROWS_PER_BLOCK);                  // 128 i32

  const int t   = threadIdx.x;
  const int p0  = blockIdx.x * ROWS_PER_BLOCK;
  const int bat = p0 >> 12;          // 4096 pixels per batch image (128 | 4096)
  const int hw0 = p0 & 4095;

  // ---- stage x tile: NCHW global -> [row][d] bf16 LDS; b128 loads along hw ----
  {
    const int pl4 = (t & 31) * 4;
    const size_t base = ((size_t)bat * DIM) * 4096 + (size_t)hw0 + pl4;
    for (int d = t >> 5; d < DIM; d += 8) {
      f32x4 v = *(const f32x4*)(x + base + (size_t)d * 4096);
      xa[(pl4 + 0) * LDS_STRIDE + d] = f2bf(v.x);
      xa[(pl4 + 1) * LDS_STRIDE + d] = f2bf(v.y);
      xa[(pl4 + 2) * LDS_STRIDE + d] = f2bf(v.z);
      xa[(pl4 + 3) * LDS_STRIDE + d] = f2bf(v.w);
    }
  }

  const int lane    = t & 31;
  const int wave    = t >> 5;
  const int rowBase = wave * 16;     // this wave's 16-row slice of the block tile
  const int lx      = lane & 15;
  const int hi      = lane >> 4;

  float bestD[8];
  int   bestI[8];
  #pragma unroll
  for (int v = 0; v < 8; ++v) { bestD[v] = 3.4e38f; bestI[v] = 0; }

  for (int chunk = 0; chunk < 4; ++chunk) {
    __syncthreads();                              // xa ready / prev compute done
    if (t < ROWS_PER_BLOCK) wn[t] = 0.0f;
    __syncthreads();

    // stage 128 codewords fp32->bf16, accumulate ||w||^2 in fp32
    for (int i = t; i < (ROWS_PER_BLOCK * DIM) / 4; i += 256) {
      const int e   = i * 4;
      const int row = e >> 8;
      const int col = e & 255;
      f32x4 wv = *(const f32x4*)(weight + (size_t)(chunk * 128 + row) * DIM + col);
      unsigned int lo = (unsigned int)f2bf(wv.x) | ((unsigned int)f2bf(wv.y) << 16);
      unsigned int hh = (unsigned int)f2bf(wv.z) | ((unsigned int)f2bf(wv.w) << 16);
      unsigned int* dst = (unsigned int*)(wb + row * LDS_STRIDE + col);
      dst[0] = lo; dst[1] = hh;
      atomicAdd(&wn[row], wv.x * wv.x + wv.y * wv.y + wv.z * wv.z + wv.w * wv.w);
    }
    __syncthreads();

    v8f acc[8];
    const v8f zero = {0.f, 0.f, 0.f, 0.f, 0.f, 0.f, 0.f, 0.f};
    #pragma unroll
    for (int ct = 0; ct < 8; ++ct) acc[ct] = zero;

    #pragma unroll
    for (int ks = 0; ks < 8; ++ks) {
      // A fragment (16x32 bf16): lane row = lx, K chunks at hi*8 and hi*8+16
      FragAB a;
      const unsigned short* pa = xa + (rowBase + lx) * LDS_STRIDE + ks * 32 + hi * 8;
      a.q[0] = *(const u32x4*)(pa);
      a.q[1] = *(const u32x4*)(pa + 16);
      // preload ALL 8 B fragments first so waits can stagger (overlap DS with XDL)
      FragAB bfArr[8];
      const int kbB = ks * 32 + hi * 8;
      #pragma unroll
      for (int ct = 0; ct < 8; ++ct) {
        const unsigned short* pb = wb + (ct * 16 + lx) * LDS_STRIDE + kbB;
        bfArr[ct].q[0] = *(const u32x4*)(pb);
        bfArr[ct].q[1] = *(const u32x4*)(pb + 16);
      }
      #pragma unroll
      for (int ct = 0; ct < 8; ++ct)
        acc[ct] = __builtin_amdgcn_wmma_f32_16x16x32_bf16(
            false, a.v, false, bfArr[ct].v, (short)0, acc[ct], false, false);
    }

    // fold: dist = ||w||^2 - 2*dot  (||f||^2 is row-constant, irrelevant to argmin)
    #pragma unroll
    for (int ct = 0; ct < 8; ++ct) {
      const float w2  = wn[ct * 16 + lx];
      const int   col = chunk * 128 + ct * 16 + lx;
      #pragma unroll
      for (int v = 0; v < 8; ++v) {
        const float d = w2 - 2.0f * acc[ct][v];
        if (d < bestD[v]) { bestD[v] = d; bestI[v] = col; }
      }
    }
  }

  // cross-lane argmin within each 16-lane group (C layout: lane = N, vgpr = M/M+8)
  #pragma unroll
  for (int m = 1; m <= 8; m <<= 1) {
    #pragma unroll
    for (int v = 0; v < 8; ++v) {
      const float od = __shfl_xor(bestD[v], m, 32);
      const int   oi = __shfl_xor(bestI[v], m, 32);
      if (od < bestD[v] || (od == bestD[v] && oi < bestI[v])) { bestD[v] = od; bestI[v] = oi; }
    }
  }
  if (lx == 0) {
    #pragma unroll
    for (int v = 0; v < 8; ++v) ids[rowBase + hi * 8 + v] = bestI[v];
  }
  __syncthreads();

  // ---- outputs: [loss(1)][quantized NCHW 16.7M][encodings 33.5M][idx 65536] ----
  float* out_loss = out;
  float* out_q    = out + 1;
  float* out_enc  = out + 1 + (size_t)N_ROWS * DIM;
  float* out_idx  = out_enc + (size_t)N_ROWS * KCODES;

  if (t < ROWS_PER_BLOCK) out_idx[p0 + t] = (float)ids[t];

  // encodings: zero (b64, region starts at odd float offset -> shift by 1) + ones
  float* encBase = out_enc + (size_t)p0 * KCODES;
  if (t == 0) { encBase[0] = 0.0f; encBase[ROWS_PER_BLOCK * KCODES - 1] = 0.0f; }
  f32x2* e2 = (f32x2*)(encBase + 1);               // 8-byte aligned
  for (int i = t; i < (ROWS_PER_BLOCK * KCODES - 2) / 2; i += 256)
    e2[i] = (f32x2){0.f, 0.f};
  __syncthreads();
  if (t < ROWS_PER_BLOCK) encBase[(size_t)t * KCODES + ids[t]] = 1.0f;

  // quantized gather (b128 stores, NCHW-coalesced); loss = vq + commit = 2 * MSE
  float lsum = 0.0f;
  {
    const int pl4 = (t & 31) * 4;
    const int k0 = ids[pl4 + 0], k1 = ids[pl4 + 1];
    const int k2 = ids[pl4 + 2], k3 = ids[pl4 + 3];
    const size_t base = ((size_t)bat * DIM) * 4096 + (size_t)hw0 + pl4;
    for (int d = t >> 5; d < DIM; d += 8) {
      f32x4 wv;
      wv.x = weight[(size_t)k0 * DIM + d];
      wv.y = weight[(size_t)k1 * DIM + d];
      wv.z = weight[(size_t)k2 * DIM + d];
      wv.w = weight[(size_t)k3 * DIM + d];
      const size_t off = base + (size_t)d * 4096;
      f32x4 xv = *(const f32x4*)(x + off);
      *(f32x4*)(out_q + off) = wv;
      f32x4 df = wv - xv;
      lsum += df.x * df.x + df.y * df.y + df.z * df.z + df.w * df.w;
    }
  }
  #pragma unroll
  for (int m = 16; m >= 1; m >>= 1) lsum += __shfl_xor(lsum, m, 32);
  if (lane == 0) atomicAdd(out_loss, lsum * (2.0f / 16777216.0f));
}

extern "C" void kernel_launch(void* const* d_in, const int* in_sizes, int n_in,
                              void* d_out, int out_size, void* d_ws, size_t ws_size,
                              hipStream_t stream) {
  const float* x = (const float*)d_in[0];
  const float* w = (const float*)d_in[1];
  float* out = (float*)d_out;

  vq_zero_loss<<<1, 1, 0, stream>>>(out);

  const size_t shmem = (size_t)2 * ROWS_PER_BLOCK * LDS_STRIDE * sizeof(unsigned short)
                     + ROWS_PER_BLOCK * sizeof(float)
                     + ROWS_PER_BLOCK * sizeof(int);        // 136,192 B (<320 KB WGP LDS)
  vq_main<<<N_ROWS / ROWS_PER_BLOCK, 256, shmem, stream>>>(x, w, out);
}
